// GraphCell_33741263077902
// MI455X (gfx1250) — matrix operand order repaired and checked by reading, not compiled, not run on hardware
//
#include <hip/hip_runtime.h>

#define NN 50000
#define DIN 256

typedef __attribute__((ext_vector_type(16))) __bf16 bf16x16;
typedef __attribute__((ext_vector_type(8)))  float  f32x8;

union frag { bf16x16 v; uint4 u[2]; };

// ---------------- degree / norm ----------------
__global__ void k_deg_init(float* __restrict__ deg, int n) {
  int i = blockIdx.x * blockDim.x + threadIdx.x;
  if (i < n) deg[i] = 1.0f;            // self-loop
}

__global__ void k_deg_accum(const int* __restrict__ rows, float* __restrict__ deg, int e) {
  int i = blockIdx.x * blockDim.x + threadIdx.x;
  if (i < e) unsafeAtomicAdd(&deg[rows[i]], 1.0f);
}

__global__ void k_rsqrt_inplace(float* __restrict__ deg, int n) {
  int i = blockIdx.x * blockDim.x + threadIdx.x;
  if (i < n) deg[i] = rsqrtf(deg[i]);  // deg >= 1 always (self-loops)
}

// ---------------- input fusion: x = qe * obj -> bf16 ----------------
__global__ void k_fuse_bf16(const float* __restrict__ qe, const float* __restrict__ obj,
                            __bf16* __restrict__ xb, int total) {
  int i = blockIdx.x * blockDim.x + threadIdx.x;
  if (i < total) xb[i] = (__bf16)(qe[i] * obj[i]);
}

// ---------------- weight: W[K][Do] -> Wt[Do][K] bf16 ----------------
__global__ void k_weight_T_bf16(const float* __restrict__ W, __bf16* __restrict__ Wt,
                                int K, int Do) {
  int t = blockIdx.x * blockDim.x + threadIdx.x;
  if (t < K * Do) {
    int n = t / K, k = t - n * K;
    Wt[t] = (__bf16)W[(size_t)k * Do + n];
  }
}

// ---------------- WMMA GEMM: C[n x Do] = A[n x K](bf16) * Wt^T ----------------
// A row-major (n x K), Bt = W^T row-major (Do x K).
// One wave computes a 16x64 output slab (4 n-tiles). A fragment is loaded once
// per K-step and reused across 4 WMMAs. Fragments are software-pipelined:
// the next B fragment (and next K-step's A fragment) is prefetched before the
// WMMA consuming the current ones, so waits on loadcnt are partial.
__global__ void k_gemm_wmma_bf16(const __bf16* __restrict__ A,
                                 const __bf16* __restrict__ Bt,
                                 float* __restrict__ C,
                                 int nrows, int K, int Do) {
  int wave = threadIdx.x >> 5;
  int lane = threadIdx.x & 31;
  int ngrp = Do >> 6;                       // groups of 4 n-tiles (64 cols)
  int wid  = blockIdx.x * (blockDim.x >> 5) + wave;
  int total = (nrows >> 4) * ngrp;
  if (wid >= total) return;                 // wave-uniform: EXEC stays all-ones
  int tm = wid / ngrp;
  int tg = wid - tm * ngrp;

  int m    = (tm << 4) + (lane & 15);
  int n0   = (tg << 6) + (lane & 15);       // first of 4 column tiles
  int asel = (lane & 16) ? 8 : 0;           // A: K sub-block per lane-half
  int bsel = (lane & 16) ? 8 : 0;           // B: K sub-block per lane-half

  const __bf16* arow = A  + (size_t)m * K;
  const __bf16* brow = Bt + (size_t)n0 * K; // tile j is at + j*16*K

  f32x8 acc[4] = {};
  frag a0, a1, b0, b1;

  // prologue: first A fragment and first B fragment
  a0.u[0] = *(const uint4*)(arow + asel);
  a0.u[1] = *(const uint4*)(arow + 16 + asel);
  b0.u[0] = *(const uint4*)(brow + bsel);
  b0.u[1] = *(const uint4*)(brow + bsel + 16);

  const int KSTEPS = K >> 5;
  int kk = 0;
  for (int s = 0; s < KSTEPS; ++s, kk += 32) {
    int kkn = (s + 1 < KSTEPS) ? kk + 32 : kk;   // clamped (in-bounds) prefetch
    // prefetch next K-step's A fragment
    a1.u[0] = *(const uint4*)(arow + kkn + asel);
    a1.u[1] = *(const uint4*)(arow + kkn + 16 + asel);
    #pragma unroll
    for (int j = 0; j < 4; ++j) {
      // prefetch next B fragment: tile j+1 at this kk, or tile 0 at next kk
      const __bf16* bp = (j < 3) ? (brow + (size_t)(j + 1) * 16 * K + kk + bsel)
                                 : (brow + kkn + bsel);
      b1.u[0] = *(const uint4*)(bp);
      b1.u[1] = *(const uint4*)(bp + 16);
      acc[j] = __builtin_amdgcn_wmma_f32_16x16x32_bf16(
                   /*neg_a=*/false, a0.v, /*neg_b=*/false, b0.v,
                   /*c_mod=*/(short)0, acc[j],
                   /*reuse_a=*/false, /*reuse_b=*/false);
      b0 = b1;
    }
    a0 = a1;
  }

  int mbase = (tm << 4) + ((lane & 16) ? 8 : 0);
  #pragma unroll
  for (int j = 0; j < 4; ++j) {
    int ncol = n0 + j * 16;                 // n0 already carries lane&15
    #pragma unroll
    for (int r = 0; r < 8; ++r)
      C[(size_t)(mbase + r) * Do + ncol] = acc[j][r];
  }
}

// ---------------- self-loop term: acc = dinv[i]^2 * h ----------------
__global__ void k_selfloop(const float* __restrict__ dinv, const float* __restrict__ h,
                           float* __restrict__ acc, int total, int d) {
  int idx = blockIdx.x * blockDim.x + threadIdx.x;
  if (idx < total) {
    int i = idx / d;
    float w = dinv[i];
    acc[idx] = w * w * h[idx];
  }
}

// ---------------- edge scatter: acc[r] += dinv[r]*dinv[c]*h[c]; wave/edge ----------------
__global__ void k_edge_scatter(const int* __restrict__ ei, int E,
                               const float* __restrict__ dinv,
                               const float* __restrict__ h,
                               float* __restrict__ acc, int d) {
  int wave = threadIdx.x >> 5;
  int lane = threadIdx.x & 31;
  int e = blockIdx.x * (blockDim.x >> 5) + wave;
  if (e >= E) return;
  e = __builtin_amdgcn_readfirstlane(e);              // provably uniform -> SMEM path
  int r = __builtin_amdgcn_readfirstlane(ei[e]);      // destination
  int c = __builtin_amdgcn_readfirstlane(ei[E + e]);  // source
  float w = dinv[r] * dinv[c];                        // uniform scalar loads
  const float* hs = h   + (size_t)c * d;
  float*       ad = acc + (size_t)r * d;
  for (int f = lane; f < d; f += 32)
    unsafeAtomicAdd(&ad[f], w * hs[f]);
}

// ---------------- epilogues ----------------
__global__ void k_bias_relu_bf16(const float* __restrict__ acc, const float* __restrict__ b,
                                 __bf16* __restrict__ xb, int total, int d) {
  int idx = blockIdx.x * blockDim.x + threadIdx.x;
  if (idx < total) {
    int f = idx % d;
    float v = acc[idx] + b[f];
    xb[idx] = (__bf16)(v > 0.0f ? v : 0.0f);
  }
}

__global__ void k_bias_out(const float* __restrict__ acc, const float* __restrict__ b,
                           float* __restrict__ out, int total, int d) {
  int idx = blockIdx.x * blockDim.x + threadIdx.x;
  if (idx < total) {
    int f = idx % d;
    out[idx] = acc[idx] + b[f];
  }
}

// ---------------- host ----------------
extern "C" void kernel_launch(void* const* d_in, const int* in_sizes, int n_in,
                              void* d_out, int out_size, void* d_ws, size_t ws_size,
                              hipStream_t stream) {
  (void)n_in; (void)out_size; (void)ws_size;
  const float* qe  = (const float*)d_in[0];
  const float* obj = (const float*)d_in[1];
  const int*   ei  = (const int*)d_in[2];
  const float* Ws[3] = { (const float*)d_in[3], (const float*)d_in[5], (const float*)d_in[7] };
  const float* bs[3] = { (const float*)d_in[4], (const float*)d_in[6], (const float*)d_in[8] };

  const int n = in_sizes[0] / DIN;   // 50000
  const int E = in_sizes[2] / 2;     // 800000

  // workspace carve-out (256B aligned), total ~128.3 MB — hot set is L2-resident
  char* ws = (char*)d_ws;
  size_t off = 0;
  auto carve = [&](size_t bytes) -> char* {
    char* p = ws + off;
    off = (off + bytes + 255) & ~(size_t)255;
    return p;
  };
  float*  dinv = (float*)carve((size_t)n * 4);
  __bf16* xb   = (__bf16*)carve((size_t)n * 256 * 2);
  float*  h    = (float*)carve((size_t)n * 256 * 4);
  float*  acc  = (float*)carve((size_t)n * 256 * 4);
  __bf16* Wt   = (__bf16*)carve((size_t)256 * 256 * 2);

  const int TB = 256;

  // normalization: deg = 1 + indeg; dinv = rsqrt(deg)
  k_deg_init<<<(n + TB - 1) / TB, TB, 0, stream>>>(dinv, n);
  k_deg_accum<<<(E + TB - 1) / TB, TB, 0, stream>>>(ei, dinv, E);
  k_rsqrt_inplace<<<(n + TB - 1) / TB, TB, 0, stream>>>(dinv, n);

  // x = qe * obj -> bf16
  int tot_in = n * DIN;
  k_fuse_bf16<<<(tot_in + TB - 1) / TB, TB, 0, stream>>>(qe, obj, xb, tot_in);

  const int dims_out[3] = { 256, 256, 128 };
  for (int l = 0; l < 3; ++l) {
    const int K = 256, Do = dims_out[l];

    k_weight_T_bf16<<<(K * Do + TB - 1) / TB, TB, 0, stream>>>(Ws[l], Wt, K, Do);

    int waves = (n >> 4) * (Do >> 6);   // one wave per 16x64 output slab
    k_gemm_wmma_bf16<<<(waves + 7) / 8, TB, 0, stream>>>(xb, Wt, h, n, K, Do);

    int nd = n * Do;
    k_selfloop<<<(nd + TB - 1) / TB, TB, 0, stream>>>(dinv, h, acc, nd, Do);
    k_edge_scatter<<<(E + 7) / 8, TB, 0, stream>>>(ei, E, dinv, h, acc, Do);

    if (l < 2)
      k_bias_relu_bf16<<<(nd + TB - 1) / TB, TB, 0, stream>>>(acc, bs[l], xb, nd, Do);
    else
      k_bias_out<<<(nd + TB - 1) / TB, TB, 0, stream>>>(acc, bs[l], (float*)d_out, nd, Do);
  }
}
